// Decoder_19774029431674
// MI455X (gfx1250) — matrix operand order repaired and checked by reading, not compiled, not run on hardware
//
#include <hip/hip_runtime.h>

typedef __attribute__((ext_vector_type(16))) __bf16        v16bf;
typedef __attribute__((ext_vector_type(8)))  float         v8f;
typedef __attribute__((ext_vector_type(4)))  unsigned int  v4u;

union Frag { v16bf v; unsigned int u[8]; v4u q[2]; };

__device__ __forceinline__ unsigned short f2bf(float f) {
  union { float f; unsigned int u; } c; c.f = f;
  return (unsigned short)((c.u + 0x7FFFu + ((c.u >> 16) & 1u)) >> 16);  // RNE
}

// ---------------- degree / norm ----------------
__global__ void init_deg(float* deg, int n) {
  int i = blockIdx.x * blockDim.x + threadIdx.x;
  if (i < n) deg[i] = 1.0f;  // self loop
}
__global__ void accum_deg(const long long* __restrict__ dst, float* deg, int e) {
  int i = blockIdx.x * blockDim.x + threadIdx.x;
  if (i < e) atomicAdd(&deg[(int)dst[i]], 1.0f);
}
__global__ void to_dinv(float* d, int n) {
  int i = blockIdx.x * blockDim.x + threadIdx.x;
  if (i < n) d[i] = rsqrtf(d[i]);  // deg >= 1 always
}

// ---------------- precision conversions (hoisted out of GEMM) ----------------
// PRE=true fuses layer-2 input op: relu(acc + b1) during the bf16 pack.
template <bool PRE>
__global__ void cvt_bf16(const float* __restrict__ in, const float* __restrict__ bias,
                         unsigned short* __restrict__ o, int total, int fmask) {
  int i = blockIdx.x * blockDim.x + threadIdx.x;
  if (i < total) {
    float v = in[i];
    if (PRE) v = fmaxf(v + bias[i & fmask], 0.f);
    o[i] = f2bf(v);
  }
}

// Wt[n*K + k] = bf16(W[k*Nc + n])  (transpose so B fragments are contiguous)
__global__ void cvt_wt(const float* __restrict__ W, unsigned short* __restrict__ Wt,
                       int total, int kshift, int kmask, int Nc) {
  int i = blockIdx.x * blockDim.x + threadIdx.x;
  if (i < total) {
    int n = i >> kshift, k = i & kmask;
    Wt[i] = f2bf(W[(size_t)k * Nc + n]);
  }
}

// ---------------- WMMA GEMM: C[M x Nc] = A[M x K] @ Wt^T, bf16 in / f32 acc ---
// One wave per 16x16 tile. A frag (16-bit A 16x32): lanes 0-15 K 0-7 & 16-23,
// lanes 16-31 K 8-15 & 24-31 -> two b128 loads at +ahalf, +ahalf+16.
// B frag (32x16): lanes 0-15 K 0-7 & 16-23, lanes 16-31 K 8-15 & 24-31 -> two b128 loads.
__global__ void gemm_wmma(const unsigned short* __restrict__ A,
                          const unsigned short* __restrict__ Bt,
                          float* __restrict__ C, int K, int Nc, int totalWaves) {
  int wv = (int)((blockIdx.x * blockDim.x + threadIdx.x) >> 5);
  if (wv >= totalWaves) return;  // wave-uniform branch: EXEC stays all-1s
  int lane = threadIdx.x & 31;
  int tilesN = Nc >> 4;
  int tm = wv / tilesN;
  int tn = wv - tm * tilesN;

  int l15   = lane & 15;
  int hi    = lane >> 4;      // 0 or 1
  int ahalf = hi * 8;         // A K-offset inside 32-block
  int bko   = hi * 8;         // B K-offset inside 32-block

  const unsigned short* ap = A  + (size_t)(tm * 16 + l15) * (size_t)K + ahalf;
  const unsigned short* bp = Bt + (size_t)(tn * 16 + l15) * (size_t)K + bko;
  v8f acc = {};

  for (int kk = 0; kk < K; kk += 32) {
    Frag a, b;
    a.q[0] = *(const v4u*)(ap + kk);        // K = ahalf + 0..7
    a.q[1] = *(const v4u*)(ap + kk + 16);   // K = ahalf + 16..23
    b.q[0] = *(const v4u*)(bp + kk);        // K = bko + 0..7
    b.q[1] = *(const v4u*)(bp + kk + 16);   // K = bko + 16..23
    acc = __builtin_amdgcn_wmma_f32_16x16x32_bf16(
        /*neg_a=*/false, a.v, /*neg_b=*/false, b.v,
        /*c_mod=*/(short)0, acc, /*reuse_a=*/false, /*reuse_b=*/false);
  }

  // C layout: VGPR r -> row (hi*8 + r), col = lane&15
  float* cp = C + (size_t)(tm * 16 + hi * 8) * (size_t)Nc + tn * 16 + l15;
#pragma unroll
  for (int r = 0; r < 8; ++r) cp[(size_t)r * Nc] = acc[r];
}

// ---------------- scatter (self-loop init + edge atomics, L2-resident) -------
__global__ void self_init(const float* __restrict__ h, const float* __restrict__ dinv,
                          float* o, int total, int shf) {
  int i = blockIdx.x * blockDim.x + threadIdx.x;
  if (i < total) {
    float dv = dinv[i >> shf];
    o[i] = h[i] * dv * dv;
  }
}

__global__ void edge_scatter(const float* __restrict__ h, const float* __restrict__ dinv,
                             const long long* __restrict__ src, const long long* __restrict__ dst,
                             float* o, int E, int F) {
  int gw = (int)((blockIdx.x * blockDim.x + threadIdx.x) >> 5);
  int lane = threadIdx.x & 31;
  if (gw >= E) return;
  int s = (int)src[gw];
  int d = (int)dst[gw];
  float nrm = dinv[s] * dinv[d];
  const float* hs = h + (size_t)s * F;
  float* od = o + (size_t)d * F;
  for (int f = lane; f < F; f += 32)
    atomicAdd(&od[f], hs[f] * nrm);
}

__global__ void bias_relu(float* o, const float* __restrict__ b, int total, int fmask) {
  int i = blockIdx.x * blockDim.x + threadIdx.x;
  if (i < total) o[i] = fmaxf(o[i] + b[i & fmask], 0.f);
}

extern "C" void kernel_launch(void* const* d_in, const int* in_sizes, int n_in,
                              void* d_out, int out_size, void* d_ws, size_t ws_size,
                              hipStream_t stream) {
  const float*     x   = (const float*)d_in[0];
  const long long* ei  = (const long long*)d_in[1];   // int64 [2, E]
  const float*     W1  = (const float*)d_in[2];
  const float*     b1  = (const float*)d_in[3];
  const float*     W2  = (const float*)d_in[4];
  const float*     b2  = (const float*)d_in[5];
  float*           out = (float*)d_out;

  const int HID = in_sizes[3];            // 256
  const int OUT = in_sizes[5];            // 64
  const int IN  = in_sizes[2] / HID;      // 128
  const int N   = in_sizes[0] / IN;       // 50000
  const int E   = in_sizes[1] / 2;        // 800000
  const long long* srcp = ei;
  const long long* dstp = ei + E;

  // workspace layout (floats); bf16 buffers alias dead f32 regions
  float* ws   = (float*)d_ws;
  float* dinv = ws;                                   // N floats
  float* h0   = ws + N;                               // N*HID floats
  float* acc1 = h0 + (size_t)N * HID;                 // N*HID floats
  unsigned short* w1t = (unsigned short*)(acc1 + (size_t)N * HID);  // HID*IN bf16
  unsigned short* w2t = w1t + (size_t)HID * IN;                     // OUT*HID bf16
  unsigned short* xb  = (unsigned short*)acc1;        // N*IN bf16 (dead after GEMM1)
  unsigned short* a2b = (unsigned short*)h0;          // N*HID bf16 (h0 dead by then)
  float* h2 = h0 + (size_t)N * HID / 2;               // N*OUT floats, after a2b

  // 1) symmetric GCN normalization: dinv = rsqrt(1 + indegree)
  init_deg <<<(N + 255) / 256, 256, 0, stream>>>(dinv, N);
  accum_deg<<<(E + 255) / 256, 256, 0, stream>>>(dstp, dinv, E);
  to_dinv  <<<(N + 255) / 256, 256, 0, stream>>>(dinv, N);

  // 2) convert operands once, then h0 = x @ W1 via WMMA
  {
    int tot = N * IN;
    cvt_bf16<false><<<(tot + 255) / 256, 256, 0, stream>>>(x, nullptr, xb, tot, 0);
    int wtot = IN * HID;
    cvt_wt<<<(wtot + 255) / 256, 256, 0, stream>>>(W1, w1t, wtot, __builtin_ctz(IN), IN - 1, HID);
    int waves = (N / 16) * (HID / 16);
    gemm_wmma<<<(waves + 7) / 8, 256, 0, stream>>>(xb, w1t, h0, IN, HID, waves);
  }

  // 3) acc1 = dinv^2 * h0 (self loops) + sum_e norm_e * h0[src]
  {
    int shf = __builtin_ctz(HID);
    int tot = N * HID;
    self_init   <<<(tot + 255) / 256, 256, 0, stream>>>(h0, dinv, acc1, tot, shf);
    edge_scatter<<<(E * 32 + 255) / 256, 256, 0, stream>>>(h0, dinv, srcp, dstp, acc1, E, HID);
  }

  // 4) a2b = bf16(relu(acc1 + b1));  h2 = a2b @ W2 via WMMA
  {
    int tot = N * HID;
    cvt_bf16<true><<<(tot + 255) / 256, 256, 0, stream>>>(acc1, b1, a2b, tot, HID - 1);
    int wtot = HID * OUT;
    cvt_wt<<<(wtot + 255) / 256, 256, 0, stream>>>(W2, w2t, wtot, __builtin_ctz(HID), HID - 1, OUT);
    int waves = (N / 16) * (OUT / 16);
    gemm_wmma<<<(waves + 7) / 8, 256, 0, stream>>>(a2b, w2t, h2, HID, OUT, waves);
  }

  // 5) out = relu(scatter(h2) + b2)
  {
    int shf = __builtin_ctz(OUT);
    int tot = N * OUT;
    self_init   <<<(tot + 255) / 256, 256, 0, stream>>>(h2, dinv, out, tot, shf);
    edge_scatter<<<(E * 32 + 255) / 256, 256, 0, stream>>>(h2, dinv, srcp, dstp, out, E, OUT);
    bias_relu   <<<(tot + 255) / 256, 256, 0, stream>>>(out, b2, tot, OUT - 1);
  }
}
